// TemporalAttention_61710090109624
// MI455X (gfx1250) — compile-verified, matrix operand order checked by reading, not run on hardware
//
#include <hip/hip_runtime.h>

typedef __attribute__((ext_vector_type(16))) _Float16 v16h;
typedef __attribute__((ext_vector_type(8)))  _Float16 v8h;
typedef __attribute__((ext_vector_type(8)))  float    v8f;
typedef __attribute__((ext_vector_type(4)))  float    v4f;

#define DM    1024      // model dim D
#define MTOT  32768     // B*T*N tokens
#define BB    4
#define TT    64
#define NVERT 128
#define NHEAD 32

__device__ __forceinline__ v16h frag_ld(const _Float16* base, int stride,
                                        int row0, int col0, int lane) {
  // 16-bit A/B fragment, wave32: lane holds row (row0 + lane&15),
  // halves {kb..kb+7} and {kb+16..kb+23}, kb = (lane<16)?0:8.
  const int m  = lane & 15;
  const int kb = (lane & 16) ? 8 : 0;
  const _Float16* p = base + (size_t)(row0 + m) * stride + col0 + kb;
  v8h lo = *(const v8h*)(p);
  v8h hi = *(const v8h*)(p + 16);
  v16h r;
#pragma unroll
  for (int i = 0; i < 8; ++i) { r[i] = lo[i]; r[8 + i] = hi[i]; }
  return r;
}

__device__ __forceinline__ v8f wmma16(v16h a, v16h b, v8f c) {
  return __builtin_amdgcn_wmma_f32_16x16x32_f16(false, a, false, b,
                                                (short)0, c, false, false);
}

// Async global->LDS copy (CDNA5): 16B per lane, tracked by ASYNCcnt.
// vdst VGPR carries the per-lane LDS byte offset; GV addressing mode.
__device__ __forceinline__ void async_ld_b128(unsigned lds_off, const void* gaddr) {
  asm volatile("global_load_async_to_lds_b128 %0, %1, off"
               :: "v"(lds_off), "v"(gaddr) : "memory");
}
__device__ __forceinline__ void wait_asynccnt0() {
  asm volatile("s_wait_asynccnt 0x0" ::: "memory");
}

// ---------------------------------------------------------------------------
// Fused GEMM + bias + ReLU:  out = relu(A @ W + bias)
//   CONCAT_A=true : A = concat(X,STE) in f32 [M,2048], converted to f16 on stage
//   CONCAT_A=false: A = f16 [M,1024]
//   OUT_HALF=true : store f16 (QKV), else f32 (final output)
// Tile 128x128xBK32, 256 threads = 8 waves, each wave 32x64 (8 wmma accs).
// ---------------------------------------------------------------------------
template <bool CONCAT_A, bool OUT_HALF>
__global__ void __launch_bounds__(256)
fused_linear_relu(const float* __restrict__ A0, const float* __restrict__ A1,
                  const _Float16* __restrict__ Ah,
                  const float* __restrict__ W, const float* __restrict__ bias,
                  _Float16* __restrict__ outH, float* __restrict__ outF,
                  int Kdim) {
  __shared__ __align__(16) _Float16 sA[2][128 * 40];   // [row][k] padded
  __shared__ __align__(16) _Float16 sB[2][128 * 40];   // [n][k]  (W transposed)

  const int bm   = blockIdx.x >> 3;        // 256 row tiles
  const int bn   = blockIdx.x & 7;         // 8 col tiles (Nout = 1024)
  const int tid  = threadIdx.x;
  const int lane = tid & 31;
  const int wave = __builtin_amdgcn_readfirstlane(tid >> 5);  // scalar wave id
  const int wm   = (wave & 3) * 32;        // wave m-offset inside tile
  const int wn   = (wave >> 2) * 64;       // wave n-offset inside tile

  // staging coordinates
  const int arow = tid >> 1, acol = (tid & 1) << 4;   // A: 16 elems/thread
  const int brow = tid >> 3, bcol = (tid & 7) << 4;   // W: 16 elems/thread
  const size_t gArow = (size_t)(bm * 128 + arow);
  const float* wcol0 = W + (size_t)(bn * 128 + bcol);

  float    aReg[16];
  _Float16 ahReg[16];
  float    wReg[16];
  const int nK = Kdim >> 5;

  auto gload = [&](int ko) {
    const int gk = ko * 32 + acol;
    if (CONCAT_A) {
      const float* src = (gk < DM) ? (A0 + gArow * DM + gk)
                                   : (A1 + gArow * DM + (gk - DM));
#pragma unroll
      for (int i = 0; i < 4; ++i) {
        v4f t = *(const v4f*)(src + i * 4);
#pragma unroll
        for (int j = 0; j < 4; ++j) aReg[4 * i + j] = t[j];
      }
    } else {
      const _Float16* src = Ah + gArow * DM + gk;
      v8h t0 = *(const v8h*)(src);
      v8h t1 = *(const v8h*)(src + 8);
#pragma unroll
      for (int i = 0; i < 8; ++i) { ahReg[i] = t0[i]; ahReg[8 + i] = t1[i]; }
    }
    const float* wsrc = wcol0 + (size_t)(ko * 32 + brow) * DM;
#pragma unroll
    for (int i = 0; i < 4; ++i) {
      v4f t = *(const v4f*)(wsrc + i * 4);
#pragma unroll
      for (int j = 0; j < 4; ++j) wReg[4 * i + j] = t[j];
    }
  };

  auto lstore = [&](int buf) {
    _Float16* pa = &sA[buf][arow * 40 + acol];
    v8h h0, h1;
    if (CONCAT_A) {
#pragma unroll
      for (int i = 0; i < 8; ++i) { h0[i] = (_Float16)aReg[i]; h1[i] = (_Float16)aReg[8 + i]; }
    } else {
#pragma unroll
      for (int i = 0; i < 8; ++i) { h0[i] = ahReg[i]; h1[i] = ahReg[8 + i]; }
    }
    *(v8h*)(pa)     = h0;
    *(v8h*)(pa + 8) = h1;
#pragma unroll
    for (int i = 0; i < 16; ++i)
      sB[buf][(bcol + i) * 40 + brow] = (_Float16)wReg[i];  // transpose W -> [n][k]
  };

  const v8f vzero = {0.f, 0.f, 0.f, 0.f, 0.f, 0.f, 0.f, 0.f};
  v8f acc[2][4];
#pragma unroll
  for (int i = 0; i < 2; ++i)
#pragma unroll
    for (int j = 0; j < 4; ++j) acc[i][j] = vzero;

  gload(0);
  lstore(0);

  for (int ko = 0; ko < nK; ++ko) {
    __syncthreads();
    if (ko + 1 < nK) gload(ko + 1);          // prefetch next tile to regs
    const int buf = ko & 1;
    v16h af[2], bf[4];
#pragma unroll
    for (int i = 0; i < 2; ++i) af[i] = frag_ld(sA[buf], 40, wm + 16 * i, 0, lane);
#pragma unroll
    for (int j = 0; j < 4; ++j) bf[j] = frag_ld(sB[buf], 40, wn + 16 * j, 0, lane);
#pragma unroll
    for (int i = 0; i < 2; ++i)
#pragma unroll
      for (int j = 0; j < 4; ++j) acc[i][j] = wmma16(af[i], bf[j], acc[i][j]);
    if (ko + 1 < nK) lstore((ko + 1) & 1);   // stage while others still compute
  }

  // epilogue: bias + relu
  const int hi = (lane >> 4) & 1;
  const int nn = lane & 15;
#pragma unroll
  for (int i = 0; i < 2; ++i)
#pragma unroll
    for (int j = 0; j < 4; ++j) {
      const int gn = bn * 128 + wn + j * 16 + nn;
      const float bv = bias[gn];
#pragma unroll
      for (int r = 0; r < 8; ++r) {
        const int gm = bm * 128 + wm + i * 16 + r + 8 * hi;
        float x = acc[i][j][r] + bv;
        x = x > 0.f ? x : 0.f;
        if (OUT_HALF) outH[(size_t)gm * DM + gn] = (_Float16)x;
        else          outF[(size_t)gm * DM + gn] = x;
      }
    }
}

// ---------------------------------------------------------------------------
// Causal temporal attention, one block per (b, vertex, head).
// T=64, head K=32. 128 threads = 4 waves; wave w owns t-rows [16w,16w+16).
// Q/K tiles staged with CDNA5 async global->LDS copies (ASYNCcnt).
// ---------------------------------------------------------------------------
__global__ void __launch_bounds__(128)
causal_head_attention(const _Float16* __restrict__ Q, const _Float16* __restrict__ K,
                      const _Float16* __restrict__ V, _Float16* __restrict__ O) {
  __shared__ __align__(16) _Float16 sQ[64 * 40];   // [t][k]
  __shared__ __align__(16) _Float16 sK[64 * 40];   // [s][k]
  __shared__ __align__(16) _Float16 sVt[32 * 72];  // [k][s]  (V transposed)
  __shared__ __align__(16) _Float16 sP[64 * 72];   // [t][s]  probabilities f16

  const int h   = blockIdx.x & (NHEAD - 1);
  const int nv  = (blockIdx.x >> 5) & (NVERT - 1);
  const int b   = blockIdx.x >> 12;
  const int tid = threadIdx.x;
  const int lane = tid & 31;
  const int w    = __builtin_amdgcn_readfirstlane(tid >> 5);  // scalar wave id

  // --- stage Q, K via async global->LDS; V via VGPRs (transposed) ----------
  {
    const int row = tid >> 1, c0 = (tid & 1) << 4;
    const size_t gb = ((size_t)(b * TT + row) * NVERT + nv) * DM + h * 32 + c0;
    const unsigned qoff = (unsigned)(uintptr_t)&sQ[row * 40 + c0];
    const unsigned koff = (unsigned)(uintptr_t)&sK[row * 40 + c0];
    async_ld_b128(qoff,      Q + gb);
    async_ld_b128(qoff + 16, Q + gb + 8);
    async_ld_b128(koff,      K + gb);
    async_ld_b128(koff + 16, K + gb + 8);

    v8h v0 = *(const v8h*)(V + gb), v1 = *(const v8h*)(V + gb + 8);
#pragma unroll
    for (int i = 0; i < 8; ++i) {
      sVt[(c0 + i)     * 72 + row] = v0[i];
      sVt[(c0 + 8 + i) * 72 + row] = v1[i];
    }
    wait_asynccnt0();              // our async copies landed in LDS
  }
  __syncthreads();                 // everyone's copies landed

  // --- S = Q K^T, one wmma per 16x16 tile (skip fully-masked si > w) -------
  const int hi = (lane >> 4) & 1;
  const int nn = lane & 15;
  const v8f vzero = {0.f, 0.f, 0.f, 0.f, 0.f, 0.f, 0.f, 0.f};
  v8f sc[4];
  {
    v16h qa = frag_ld(sQ, 40, w * 16, 0, lane);
#pragma unroll
    for (int si = 0; si < 4; ++si)
      if (si <= w) {               // scalar branch: w is readfirstlane'd
        v16h kf = frag_ld(sK, 40, si * 16, 0, lane);
        sc[si] = wmma16(qa, kf, vzero);
      }
  }

  // --- scale, causal mask, row softmax (rows split across 16-lane halves) --
  const float scale = 0.17677669529663687f;            // 1/sqrt(32)
  float rmax[8], rsum[8], rinv[8];
#pragma unroll
  for (int r = 0; r < 8; ++r) rmax[r] = -3.0e38f;
#pragma unroll
  for (int si = 0; si < 4; ++si)
    if (si <= w) {
#pragma unroll
      for (int r = 0; r < 8; ++r) {
        float x = sc[si][r] * scale;
        const int trow = w * 16 + r + 8 * hi;          // C-layout: m = r + 8*(lane>=16)
        const int srow = si * 16 + nn;                 //           n = lane & 15
        if (srow > trow) x = -3.0e38f;                 // causal mask
        sc[si][r] = x;
        rmax[r] = fmaxf(rmax[r], x);
      }
    }
#pragma unroll
  for (int r = 0; r < 8; ++r) {
#pragma unroll
    for (int m = 1; m <= 8; m <<= 1)
      rmax[r] = fmaxf(rmax[r], __shfl_xor(rmax[r], m, 32));
    rsum[r] = 0.f;
  }
#pragma unroll
  for (int si = 0; si < 4; ++si)
    if (si <= w) {
#pragma unroll
      for (int r = 0; r < 8; ++r) {
        float e = __expf(sc[si][r] - rmax[r]);
        sc[si][r] = e;
        rsum[r] += e;
      }
    }
#pragma unroll
  for (int r = 0; r < 8; ++r) {
#pragma unroll
    for (int m = 1; m <= 8; m <<= 1)
      rsum[r] += __shfl_xor(rsum[r], m, 32);
    rinv[r] = __builtin_amdgcn_rcpf(rsum[r]);
  }

  // --- P to LDS (masked tiles are exact zeros) -----------------------------
#pragma unroll
  for (int si = 0; si < 4; ++si)
#pragma unroll
    for (int r = 0; r < 8; ++r) {
      _Float16 pv = (si <= w) ? (_Float16)(sc[si][r] * rinv[r]) : (_Float16)0.f;
      sP[(w * 16 + r + 8 * hi) * 72 + si * 16 + nn] = pv;
    }
  __syncthreads();

  // --- O = P V  (K-dim = 64 over s: 2 wmma steps x 2 output n-tiles) -------
  v8f o0 = vzero, o1 = vzero;
#pragma unroll
  for (int ks = 0; ks < 2; ++ks) {
    v16h pa  = frag_ld(sP, 72, w * 16, ks * 32, lane);
    v16h vb0 = frag_ld(sVt, 72, 0,  ks * 32, lane);
    v16h vb1 = frag_ld(sVt, 72, 16, ks * 32, lane);
    o0 = wmma16(pa, vb0, o0);
    o1 = wmma16(pa, vb1, o1);
  }
#pragma unroll
  for (int r = 0; r < 8; ++r) {
    const int t = w * 16 + r + 8 * hi;
    const size_t ob = ((size_t)(b * TT + t) * NVERT + nv) * DM + h * 32;
    O[ob + nn]      = (_Float16)o0[r];
    O[ob + 16 + nn] = (_Float16)o1[r];
  }
}

// ---------------------------------------------------------------------------
extern "C" void kernel_launch(void* const* d_in, const int* in_sizes, int n_in,
                              void* d_out, int out_size, void* d_ws, size_t ws_size,
                              hipStream_t stream) {
  (void)in_sizes; (void)n_in; (void)out_size; (void)ws_size;
  const float* X   = (const float*)d_in[0];
  const float* STE = (const float*)d_in[1];
  const float* Wq  = (const float*)d_in[2];
  const float* bq  = (const float*)d_in[3];
  const float* Wk  = (const float*)d_in[4];
  const float* bk  = (const float*)d_in[5];
  const float* Wv  = (const float*)d_in[6];
  const float* bv  = (const float*)d_in[7];
  const float* Wo  = (const float*)d_in[8];
  const float* bo  = (const float*)d_in[9];
  float* out = (float*)d_out;

  const size_t MD = (size_t)MTOT * DM;
  _Float16* qh = (_Float16*)d_ws;     // 64 MB each
  _Float16* kh = qh + MD;
  _Float16* vh = kh + MD;
  _Float16* ah = vh + MD;

  dim3 gg(2048), tb(256);             // (32768/128) * (1024/128)
  fused_linear_relu<true,  true ><<<gg, tb, 0, stream>>>(X, STE, nullptr, Wq, bq, qh, nullptr, 2 * DM);
  fused_linear_relu<true,  true ><<<gg, tb, 0, stream>>>(X, STE, nullptr, Wk, bk, kh, nullptr, 2 * DM);
  fused_linear_relu<true,  true ><<<gg, tb, 0, stream>>>(X, STE, nullptr, Wv, bv, vh, nullptr, 2 * DM);
  causal_head_attention<<<dim3(BB * NVERT * NHEAD), dim3(128), 0, stream>>>(qh, kh, vh, ah);
  fused_linear_relu<false, false><<<gg, tb, 0, stream>>>(nullptr, nullptr, ah, Wo, bo, nullptr, out, DM);
}
